// Classifier_38568806318157
// MI455X (gfx1250) — matrix operand order repaired and checked by reading, not compiled, not run on hardware
//
#include <hip/hip_runtime.h>
#include <hip/hip_bf16.h>
#include <math.h>

typedef __attribute__((ext_vector_type(16))) _Float16 v16h;
typedef __attribute__((ext_vector_type(8)))  _Float16 v8h;
typedef __attribute__((ext_vector_type(8)))  float    v8f;

#define Bc 4
#define Tc 513
#define Dc 512
#define INNERc 1024
#define NHc 8
#define DHc 128
#define Kc 4
#define NCc 1000
#define Mrows (Bc*Tc)   // 2052

// ---------------------------------------------------------------------------
// CDNA5 async global->LDS copy (ASYNCcnt-tracked), 16 bytes per lane.
// VDST operand = wave-relative LDS byte offset = low 32 bits of the generic
// LDS pointer.
// ---------------------------------------------------------------------------
__device__ __forceinline__ void async_cp16(void* lds, const void* g)
{
    unsigned loff = (unsigned)(size_t)lds;
    asm volatile("global_load_async_to_lds_b128 %0, %1, off"
                 :: "v"(loff), "v"(g) : "memory");
}
__device__ __forceinline__ void wait_async0()
{
    asm volatile("s_wait_asynccnt 0x0" ::: "memory");
}
__device__ __forceinline__ void wait_async3()
{
    asm volatile("s_wait_asynccnt 0x3" ::: "memory");
}

// ---------------------------------------------------------------------------
// f16 GEMM via v_wmma_f32_16x16x32_f16.  C = alpha*(A@W) + bias + residual
// A:  M x K f16 row-major (row stride lda)
// Wt: N x K f16 row-major (pre-transposed weights, row stride K)
// C:  M x N f32 (ldc)
// Block: 256 threads (8 waves), tile 64(M) x 128(N), K-step 32.
// Double-buffered LDS; tiles staged with async global->LDS b128 copies
// (next tile issued before waiting on current: s_wait_asynccnt 3).
// Rows past M are staged clamped (garbage) — never stored.
// ---------------------------------------------------------------------------
__global__ __launch_bounds__(256)
void gemm_wmma_f16(const _Float16* __restrict__ A, int lda,
                   const _Float16* __restrict__ Wt,
                   const float* __restrict__ bias,
                   const float* __restrict__ residual, int ldr,
                   float* __restrict__ C, int ldc,
                   int M, int N, int K, float alpha)
{
    __shared__ __align__(16) _Float16 As[2][64][40];    // M x K tiles
    __shared__ __align__(16) _Float16 Bs[2][128][40];   // N x K tiles

    const int tid  = threadIdx.x;
    const int lane = tid & 31;
    const int wave = tid >> 5;
    const int bm0  = blockIdx.y * 64;
    const int bn0  = blockIdx.x * 128;
    const int wm   = wave & 3;            // M sub-tile 0..3
    const int wcol = wave >> 2;           // N column group 0..1
    const int lm   = lane & 15;
    const int lhi  = lane >> 4;

    // staging decomposition: 16-byte chunks
    const int sr = tid >> 2;              // row 0..63
    const int c8 = (tid & 3) * 8;         // 8-half chunk within K-step

    int grA = bm0 + sr;                   // clamped A source row
    grA = (grA < M) ? grA : (M - 1);

    v8f acc[4] = {v8f{}, v8f{}, v8f{}, v8f{}};

    // prologue: issue tile 0 (3 async b128 per thread)
    async_cp16(&As[0][sr][c8],      A  + (size_t)grA * lda + c8);
    async_cp16(&Bs[0][sr][c8],      Wt + (size_t)(bn0 + sr) * K + c8);
    async_cp16(&Bs[0][sr + 64][c8], Wt + (size_t)(bn0 + sr + 64) * K + c8);

    int buf = 0;
    for (int k0 = 0; k0 < K; k0 += 32, buf ^= 1) {
        const int kn = k0 + 32;
        const bool hasNext = kn < K;
        if (hasNext) {   // issue next tile into other buffer before waiting
            async_cp16(&As[buf ^ 1][sr][c8],      A  + (size_t)grA * lda + kn + c8);
            async_cp16(&Bs[buf ^ 1][sr][c8],      Wt + (size_t)(bn0 + sr) * K + kn + c8);
            async_cp16(&Bs[buf ^ 1][sr + 64][c8], Wt + (size_t)(bn0 + sr + 64) * K + kn + c8);
            wait_async3();               // current tile's 3 loads complete
        } else {
            wait_async0();
        }
        __syncthreads();

        // A fragment: lane holds row (wm*16+lm); halves 0-7 <- K[8*lhi..],
        // halves 8-15 <- K[16+8*lhi..]
        v16h af;
        {
            const int row = wm * 16 + lm;
            v8h lo = *(const v8h*)&As[buf][row][lhi * 8];
            v8h hi = *(const v8h*)&As[buf][row][16 + lhi * 8];
            #pragma unroll
            for (int i = 0; i < 8; ++i) { af[i] = lo[i]; af[i + 8] = hi[i]; }
        }
        // 4 B fragments + 4 WMMAs (A reuse)
        #pragma unroll
        for (int j = 0; j < 4; ++j) {
            const int rr = (wcol * 4 + j) * 16 + lm;
            v8h lo = *(const v8h*)&Bs[buf][rr][lhi * 8];
            v8h hi = *(const v8h*)&Bs[buf][rr][16 + lhi * 8];
            v16h bf;
            #pragma unroll
            for (int i = 0; i < 8; ++i) { bf[i] = lo[i]; bf[i + 8] = hi[i]; }
            acc[j] = __builtin_amdgcn_wmma_f32_16x16x32_f16(
                false, af, false, bf, (short)0, acc[j], false, false);
        }
        __syncthreads();
    }

    // epilogue: D vgpr r -> M = r + 8*lhi, N = lm
    #pragma unroll
    for (int j = 0; j < 4; ++j) {
        const int gn = bn0 + (wcol * 4 + j) * 16 + lm;
        #pragma unroll
        for (int r = 0; r < 8; ++r) {
            int gm = bm0 + wm * 16 + r + 8 * lhi;
            if (gm < M) {
                float v = alpha * acc[j][r];
                if (bias)     v += bias[gn];
                if (residual) v += residual[(size_t)gm * ldr + gn];
                C[(size_t)gm * ldc + gn] = v;
            }
        }
    }
}

// ---------------------------------------------------------------------------
// weight convert + transpose: Wt[n][k] = (f16) W[k][n]   (coalesced reads)
// ---------------------------------------------------------------------------
__global__ void wtrans_f16_kernel(const float* __restrict__ W,
                                  _Float16* __restrict__ Wt,
                                  int K, int N, int total)
{
    int idx = blockIdx.x * blockDim.x + threadIdx.x;
    if (idx >= total) return;
    int k = idx / N, n = idx % N;
    Wt[(size_t)n * K + k] = (_Float16)W[idx];
}

// ---------------------------------------------------------------------------
// concat [x ; cls] -> h  (B, Tc, D)
// ---------------------------------------------------------------------------
__global__ void concat_cls_kernel(const float* __restrict__ x,
                                  const float* __restrict__ cls,
                                  float* __restrict__ h, int total)
{
    int idx = blockIdx.x * blockDim.x + threadIdx.x;
    if (idx >= total) return;
    int d = idx % Dc;
    int t = (idx / Dc) % Tc;
    int b = idx / (Dc * Tc);
    h[idx] = (t < Tc - 1) ? x[((size_t)b * (Tc - 1) + t) * Dc + d] : cls[d];
}

// ---------------------------------------------------------------------------
// LayerNorm over D=512, f16 output (GEMM A operand)
// ---------------------------------------------------------------------------
__global__ __launch_bounds__(256)
void layernorm_kernel(const float* __restrict__ x, const float* __restrict__ s,
                      const float* __restrict__ bt, _Float16* __restrict__ y)
{
    int row = blockIdx.x, tid = threadIdx.x;
    __shared__ float r1[256], r2[256];
    size_t base = (size_t)row * Dc;
    float a0 = x[base + tid];
    float a1 = x[base + tid + 256];
    r1[tid] = a0 + a1;
    r2[tid] = a0 * a0 + a1 * a1;
    __syncthreads();
    for (int ss = 128; ss > 0; ss >>= 1) {
        if (tid < ss) { r1[tid] += r1[tid + ss]; r2[tid] += r2[tid + ss]; }
        __syncthreads();
    }
    float mu = r1[0] * (1.0f / Dc);
    float var = r2[0] * (1.0f / Dc) - mu * mu;
    float rstd = rsqrtf(var + 1e-5f);
    y[base + tid]       = (_Float16)((a0 - mu) * rstd * s[tid]       + bt[tid]);
    y[base + tid + 256] = (_Float16)((a1 - mu) * rstd * s[tid + 256] + bt[tid + 256]);
}

// ---------------------------------------------------------------------------
// causal depthwise conv (K=4) + SiLU -> xc (f16); also emits xu as f16
// ---------------------------------------------------------------------------
__global__ void conv_silu_kernel(const float* __restrict__ up,
                                 const float* __restrict__ ck,
                                 const float* __restrict__ cb,
                                 _Float16* __restrict__ xch,
                                 _Float16* __restrict__ xuh, int total)
{
    int idx = blockIdx.x * blockDim.x + threadIdx.x;
    if (idx >= total) return;
    int c = idx % INNERc;
    int t = (idx / INNERc) % Tc;
    int b = idx / (INNERc * Tc);
    float xub = up[(size_t)(b * Tc + t) * (2 * INNERc) + c];   // tap j = K-1
    float acc = cb[c] + xub * ck[c * Kc + (Kc - 1)];
    #pragma unroll
    for (int j = 0; j < Kc - 1; ++j) {
        int tt = t - (Kc - 1) + j;
        if (tt >= 0)
            acc += up[(size_t)(b * Tc + tt) * (2 * INNERc) + c] * ck[c * Kc + j];
    }
    xch[idx] = (_Float16)(acc / (1.0f + expf(-acc)));
    xuh[idx] = (_Float16)xub;
}

// ---------------------------------------------------------------------------
// gate projections: ip = xc@w_ig + b_ig ; fp = xc@w_fg + b_fg (N=8)
// ---------------------------------------------------------------------------
__global__ __launch_bounds__(256)
void gates_kernel(const _Float16* __restrict__ xch,
                  const float* __restrict__ wig, const float* __restrict__ big,
                  const float* __restrict__ wfg, const float* __restrict__ bfg,
                  float* __restrict__ ip, float* __restrict__ fp)
{
    int row = blockIdx.x, tid = threadIdx.x;
    int n = tid & 7, g = tid >> 3;
    __shared__ float rA[256], rB[256];
    float aI = 0.f, aF = 0.f;
    for (int kk = g; kk < INNERc; kk += 32) {
        float xv = (float)xch[(size_t)row * INNERc + kk];
        aI += xv * wig[kk * NHc + n];
        aF += xv * wfg[kk * NHc + n];
    }
    rA[tid] = aI; rB[tid] = aF;
    __syncthreads();
    for (int s = 128; s >= 8; s >>= 1) {
        if (tid < s) { rA[tid] += rA[tid + s]; rB[tid] += rB[tid + s]; }
        __syncthreads();
    }
    if (tid < 8) {
        ip[(size_t)row * NHc + tid] = rA[tid] + big[tid];
        fp[(size_t)row * NHc + tid] = rB[tid] + bfg[tid];
    }
}

// ---------------------------------------------------------------------------
// sequential mLSTM scan. 32 blocks = (b, head); 256 threads; C (128x128) in
// registers (64 floats/thread); k/v/q staged in LDS each step; h overwrites q.
// ---------------------------------------------------------------------------
__global__ __launch_bounds__(256)
void mlstm_scan_kernel(float* __restrict__ qh,        // in: q, out: h
                       const float* __restrict__ kk,
                       const float* __restrict__ vv,
                       const float* __restrict__ ip,
                       const float* __restrict__ fp)
{
    const int b  = blockIdx.x / NHc;
    const int hh = blockIdx.x % NHc;
    const int tid  = threadIdx.x;
    const int i    = tid & 127;       // owned C row
    const int half = tid >> 7;        // column half
    const int jb   = half * 64;

    __shared__ float sq[128], sk[128], sv[128], nvec[128], red[128];
    __shared__ float partArr[256], scal[3], mstate;

    float Creg[64];
    #pragma unroll
    for (int j = 0; j < 64; ++j) Creg[j] = 0.f;
    if (tid < 128) nvec[tid] = 0.f;
    if (tid == 0)  mstate = 0.f;
    __syncthreads();

    for (int t = 0; t < Tc; ++t) {
        const size_t base = (size_t)(b * Tc + t) * INNERc + hh * DHc;
        if (tid < 128) { sq[tid] = qh[base + tid]; sv[tid] = vv[base + tid]; }
        else           { sk[tid - 128] = kk[base + tid - 128]; }
        if (tid == 0) {
            size_t gb = (size_t)(b * Tc + t) * NHc + hh;
            float fpt = fp[gb], ipt = ip[gb];
            float fl  = fminf(fpt, 0.f) - log1pf(expf(-fabsf(fpt))); // log_sigmoid
            float m   = mstate;
            float mn  = fmaxf(fl + m, ipt);
            scal[0] = expf(ipt - mn);       // input gate
            scal[1] = expf(fl + m - mn);    // forget gate
            mstate  = mn;
        }
        __syncthreads();

        float ii = scal[0], ff = scal[1];
        float vi = sv[i];
        float part = 0.f;
        #pragma unroll
        for (int j = 0; j < 64; ++j) {
            float c = ff * Creg[j] + ii * (vi * sk[jb + j]);
            Creg[j] = c;
            part += c * sq[jb + j];
        }
        partArr[tid] = part;
        if (tid < 128) {
            float nn = ff * nvec[tid] + ii * sk[tid];
            nvec[tid] = nn;
            red[tid] = nn * sq[tid];
        }
        __syncthreads();
        for (int s = 64; s > 0; s >>= 1) {
            if (tid < s) red[tid] += red[tid + s];
            __syncthreads();
        }
        if (tid == 0) scal[2] = fmaxf(fabsf(red[0]), 1.0f);
        __syncthreads();
        if (tid < 128)
            qh[base + tid] = (partArr[tid] + partArr[tid + 128]) / scal[2];
        __syncthreads();
    }
}

// ---------------------------------------------------------------------------
// per-head norm * hn_s, then * silu(z) -> hg (f16, GEMM A operand)
// ---------------------------------------------------------------------------
__global__ __launch_bounds__(128)
void hnorm_gate_kernel(const float* __restrict__ hsc,
                       const float* __restrict__ up,
                       const float* __restrict__ hns,
                       _Float16* __restrict__ hgh)
{
    int rowh = blockIdx.x;
    int hh  = rowh % NHc;
    int row = rowh / NHc;             // b*Tc + t
    int tid = threadIdx.x;
    __shared__ float r1[128], r2[128];
    size_t base = (size_t)row * INNERc + hh * DHc;
    float a = hsc[base + tid];
    r1[tid] = a; r2[tid] = a * a;
    __syncthreads();
    for (int ss = 64; ss > 0; ss >>= 1) {
        if (tid < ss) { r1[tid] += r1[tid + ss]; r2[tid] += r2[tid + ss]; }
        __syncthreads();
    }
    float mu = r1[0] * (1.0f / DHc);
    float var = r2[0] * (1.0f / DHc) - mu * mu;
    float nv = (a - mu) * rsqrtf(var + 1e-5f) * hns[hh * DHc + tid];
    float z  = up[(size_t)row * (2 * INNERc) + INNERc + hh * DHc + tid];
    float sz = z / (1.0f + expf(-z));
    hgh[base + tid] = (_Float16)(nv * sz);
}

// ---------------------------------------------------------------------------
// cls token: LN + ReLU  (B blocks, 256 threads)
// ---------------------------------------------------------------------------
__global__ __launch_bounds__(256)
void cls_ln_kernel(const float* __restrict__ h, const float* __restrict__ s,
                   const float* __restrict__ bt, float* __restrict__ clsn)
{
    int b = blockIdx.x, tid = threadIdx.x;
    __shared__ float r1[256], r2[256];
    size_t base = (size_t)(b * Tc + (Tc - 1)) * Dc;
    float a0 = h[base + tid];
    float a1 = h[base + tid + 256];
    r1[tid] = a0 + a1; r2[tid] = a0 * a0 + a1 * a1;
    __syncthreads();
    for (int ss = 128; ss > 0; ss >>= 1) {
        if (tid < ss) { r1[tid] += r1[tid + ss]; r2[tid] += r2[tid + ss]; }
        __syncthreads();
    }
    float mu = r1[0] * (1.0f / Dc);
    float var = r2[0] * (1.0f / Dc) - mu * mu;
    float rstd = rsqrtf(var + 1e-5f);
    clsn[(size_t)b * Dc + tid]       = fmaxf((a0 - mu) * rstd * s[tid]       + bt[tid],       0.f);
    clsn[(size_t)b * Dc + tid + 256] = fmaxf((a1 - mu) * rstd * s[tid + 256] + bt[tid + 256], 0.f);
}

// final FC: (B,512) @ (512,1000) + b  — trivial
__global__ void fc_kernel(const float* __restrict__ clsn,
                          const float* __restrict__ w,
                          const float* __restrict__ bias,
                          float* __restrict__ out, int total)
{
    int idx = blockIdx.x * blockDim.x + threadIdx.x;
    if (idx >= total) return;
    int b = idx / NCc, n = idx % NCc;
    float acc = bias[n];
    for (int k = 0; k < Dc; ++k)
        acc += clsn[(size_t)b * Dc + k] * w[(size_t)k * NCc + n];
    out[idx] = acc;
}

// ---------------------------------------------------------------------------
extern "C" void kernel_launch(void* const* d_in, const int* in_sizes, int n_in,
                              void* d_out, int out_size, void* d_ws, size_t ws_size,
                              hipStream_t stream)
{
    const float* x        = (const float*)d_in[0];
    const float* cls      = (const float*)d_in[1];
    const float* ln_s     = (const float*)d_in[2];
    const float* ln_b     = (const float*)d_in[3];
    const float* w_up     = (const float*)d_in[4];
    const float* b_up     = (const float*)d_in[5];
    const float* conv_k   = (const float*)d_in[6];
    const float* conv_b   = (const float*)d_in[7];
    const float* w_q      = (const float*)d_in[8];
    const float* w_k      = (const float*)d_in[9];
    const float* w_v      = (const float*)d_in[10];
    const float* w_ig     = (const float*)d_in[11];
    const float* b_ig     = (const float*)d_in[12];
    const float* w_fg     = (const float*)d_in[13];
    const float* b_fg     = (const float*)d_in[14];
    const float* hn_s     = (const float*)d_in[15];
    const float* w_down   = (const float*)d_in[16];
    const float* b_down   = (const float*)d_in[17];
    const float* fc_ln_s  = (const float*)d_in[18];
    const float* fc_ln_b  = (const float*)d_in[19];
    const float* fc_w     = (const float*)d_in[20];
    const float* fc_b     = (const float*)d_in[21];
    float* out = (float*)d_out;

    // ---- workspace carve-up: f32 region first, then f16 region (16B-aligned)
    float* ws = (float*)d_ws;
    size_t off = 0;
    float* hbuf  = ws + off; off += (size_t)Mrows * Dc;            // residual stream
    float* upbuf = ws + off; off += (size_t)Mrows * 2 * INNERc;    // up projection
    float* qbuf  = ws + off; off += (size_t)Mrows * INNERc;        // q, then h
    float* kbuf  = ws + off; off += (size_t)Mrows * INNERc;
    float* vbuf  = ws + off; off += (size_t)Mrows * INNERc;
    float* ipbuf = ws + off; off += (size_t)Mrows * NHc;
    float* fpbuf = ws + off; off += (size_t)Mrows * NHc;
    float* clsn  = ws + off; off += (size_t)Bc * Dc;

    _Float16* hws = (_Float16*)(ws + off);
    size_t hoff = 0;
    _Float16* xnh  = hws + hoff; hoff += (size_t)Mrows * Dc;        // LN out
    _Float16* xch  = hws + hoff; hoff += (size_t)Mrows * INNERc;    // conv out
    _Float16* xuh  = hws + hoff; hoff += (size_t)Mrows * INNERc;    // xu (f16)
    _Float16* hgh  = hws + hoff; hoff += (size_t)Mrows * INNERc;    // gated h
    _Float16* wupT = hws + hoff; hoff += (size_t)2 * INNERc * Dc;   // 2048 x 512
    _Float16* wqT  = hws + hoff; hoff += (size_t)INNERc * INNERc;
    _Float16* wkT  = hws + hoff; hoff += (size_t)INNERc * INNERc;
    _Float16* wvT  = hws + hoff; hoff += (size_t)INNERc * INNERc;
    _Float16* wdnT = hws + hoff; hoff += (size_t)Dc * INNERc;       // 512 x 1024
    (void)ws_size; (void)in_sizes; (void)n_in; (void)out_size;

    const float kscale = 0.08838834764831845f;   // DH^-0.5

    // concat x + cls
    {
        int total = Mrows * Dc;
        concat_cls_kernel<<<(total + 255) / 256, 256, 0, stream>>>(x, cls, hbuf, total);
    }

    const int gy = (Mrows + 63) / 64;            // 33 tile rows

    for (int blk = 0; blk < 2; ++blk) {
        const float* lns  = ln_s   + (size_t)blk * Dc;
        const float* lnb  = ln_b   + (size_t)blk * Dc;
        const float* wup  = w_up   + (size_t)blk * Dc * 2 * INNERc;
        const float* bup  = b_up   + (size_t)blk * 2 * INNERc;
        const float* ck   = conv_k + (size_t)blk * INNERc * Kc;
        const float* cb   = conv_b + (size_t)blk * INNERc;
        const float* wq   = w_q    + (size_t)blk * INNERc * INNERc;
        const float* wk   = w_k    + (size_t)blk * INNERc * INNERc;
        const float* wv   = w_v    + (size_t)blk * INNERc * INNERc;
        const float* wig  = w_ig   + (size_t)blk * INNERc * NHc;
        const float* big  = b_ig   + (size_t)blk * NHc;
        const float* wfg  = w_fg   + (size_t)blk * INNERc * NHc;
        const float* bfg  = b_fg   + (size_t)blk * NHc;
        const float* hns  = hn_s   + (size_t)blk * NHc * DHc;
        const float* wdn  = w_down + (size_t)blk * INNERc * Dc;
        const float* bdn  = b_down + (size_t)blk * Dc;

        // 0. convert + transpose this block's GEMM weights to f16 (N x K)
        {
            int t1 = Dc * 2 * INNERc;
            wtrans_f16_kernel<<<(t1 + 255) / 256, 256, 0, stream>>>(wup, wupT, Dc, 2 * INNERc, t1);
            int t2 = INNERc * INNERc;
            wtrans_f16_kernel<<<(t2 + 255) / 256, 256, 0, stream>>>(wq, wqT, INNERc, INNERc, t2);
            wtrans_f16_kernel<<<(t2 + 255) / 256, 256, 0, stream>>>(wk, wkT, INNERc, INNERc, t2);
            wtrans_f16_kernel<<<(t2 + 255) / 256, 256, 0, stream>>>(wv, wvT, INNERc, INNERc, t2);
            int t3 = INNERc * Dc;
            wtrans_f16_kernel<<<(t3 + 255) / 256, 256, 0, stream>>>(wdn, wdnT, INNERc, Dc, t3);
        }

        // 1. LayerNorm -> f16
        layernorm_kernel<<<Mrows, 256, 0, stream>>>(hbuf, lns, lnb, xnh);

        // 2. up projection (M=2052, K=512, N=2048)
        gemm_wmma_f16<<<dim3(2 * INNERc / 128, gy), 256, 0, stream>>>(
            xnh, Dc, wupT, bup, nullptr, 0,
            upbuf, 2 * INNERc, Mrows, 2 * INNERc, Dc, 1.0f);

        // 3. causal conv + SiLU -> xch (f16), xu copy -> xuh (f16)
        {
            int total = Mrows * INNERc;
            conv_silu_kernel<<<(total + 255) / 256, 256, 0, stream>>>(
                upbuf, ck, cb, xch, xuh, total);
        }

        // 4. q / k / v projections (K=1024, N=1024)
        gemm_wmma_f16<<<dim3(INNERc / 128, gy), 256, 0, stream>>>(
            xch, INNERc, wqT, nullptr, nullptr, 0,
            qbuf, INNERc, Mrows, INNERc, INNERc, 1.0f);
        gemm_wmma_f16<<<dim3(INNERc / 128, gy), 256, 0, stream>>>(
            xch, INNERc, wkT, nullptr, nullptr, 0,
            kbuf, INNERc, Mrows, INNERc, INNERc, kscale);
        gemm_wmma_f16<<<dim3(INNERc / 128, gy), 256, 0, stream>>>(
            xuh, INNERc, wvT, nullptr, nullptr, 0,
            vbuf, INNERc, Mrows, INNERc, INNERc, 1.0f);

        // gate projections
        gates_kernel<<<Mrows, 256, 0, stream>>>(xch, wig, big, wfg, bfg, ipbuf, fpbuf);

        // 5. mLSTM sequential scan (h overwrites q)
        mlstm_scan_kernel<<<Bc * NHc, 256, 0, stream>>>(qbuf, kbuf, vbuf, ipbuf, fpbuf);

        // 6. head norm + silu(z) gate -> hgh (f16)
        hnorm_gate_kernel<<<Mrows * NHc, 128, 0, stream>>>(qbuf, upbuf, hns, hgh);

        // 7. down projection + bias + residual, in-place into hbuf
        gemm_wmma_f16<<<dim3(Dc / 128, gy), 256, 0, stream>>>(
            hgh, INNERc, wdnT, bdn, hbuf, Dc,
            hbuf, Dc, Mrows, Dc, INNERc, 1.0f);
    }

    // final head
    cls_ln_kernel<<<Bc, 256, 0, stream>>>(hbuf, fc_ln_s, fc_ln_b, clsn);
    {
        int total = Bc * NCc;
        fc_kernel<<<(total + 255) / 256, 256, 0, stream>>>(clsn, fc_w, fc_b, out, total);
    }
}
